// MultiHeadSelfAttention_63891933495744
// MI455X (gfx1250) — matrix operand order repaired and checked by reading, not compiled, not run on hardware
//
#include <hip/hip_runtime.h>

#define B_DIM 4
#define T_DIM 2048
#define E_DIM 1024
#define H_DIM 16
#define D_DIM 64

typedef __attribute__((ext_vector_type(16))) _Float16 v16h;
typedef __attribute__((ext_vector_type(8)))  _Float16 v8h;
typedef __attribute__((ext_vector_type(4)))  _Float16 v4h;
typedef __attribute__((ext_vector_type(8)))  float    v8f;

union FragU { v16h v; v8h h[2]; };

// A-fragment (16x32 f16, row-major source, ld in elements).
// ISA layout: lanes 0-15 row m, V0..3 = K{0..7} pairs, V4..7 = K{16..23};
// lanes 16-31 same rows, K offset +8 within each half.
__device__ __forceinline__ v16h load_fragA(const _Float16* p, int ld) {
  const int lane = threadIdx.x & 31;
  const int m  = lane & 15;
  const int hi = lane >> 4;
  const _Float16* r = p + m * ld + hi * 8;
  FragU u;
  u.h[0] = *(const v8h*)(r);
  u.h[1] = *(const v8h*)(r + 16);
  return u.v;
}

// B-fragment (32x16 f16). Source is B^T row-major: row n holds column n of B,
// contiguous over K. ISA layout: lanes 0-15 hold K=0..7,16..23; lanes 16-31 K=8..15,24..31.
__device__ __forceinline__ v16h load_fragB(const _Float16* p, int ld) {
  const int lane = threadIdx.x & 31;
  const int nn = lane & 15;
  const int hi = lane >> 4;
  const _Float16* r = p + nn * ld + hi * 8;
  FragU u;
  u.h[0] = *(const v8h*)(r);
  u.h[1] = *(const v8h*)(r + 16);
  return u.v;
}

__device__ __forceinline__ v8f wmma16(v16h a, v16h b, v8f c) {
  return __builtin_amdgcn_wmma_f32_16x16x32_f16(false, a, false, b,
                                                (short)0, c, false, false);
}

// ---------------------------------------------------------------------------
// Kernel 1: per-head projection  out[b,h,t,d] = sum_e x[b,t,e]*W[h,e,d] + b[h,d]
// Block: 256 threads (8 waves), tile = 128 rows(t) x 64 cols(d).
// transpose_out=0 -> [B,H,T,D] f16 ; transpose_out=1 -> [B,H,D,T] f16 (for V).
// ---------------------------------------------------------------------------
__global__ __launch_bounds__(256) void qkv_proj_kernel(
    const float* __restrict__ x, const float* __restrict__ W,
    const float* __restrict__ bias, _Float16* __restrict__ out,
    int transpose_out) {
  __shared__ __align__(16) _Float16 xs[128 * 32];  // x tile (f16)
  __shared__ __align__(16) _Float16 wt[64 * 32];   // W^T tile: wt[d][e]

  const int tid = threadIdx.x;
  const int bt0 = blockIdx.x * 128;
  const int h = blockIdx.y;
  const int b = bt0 / T_DIM;
  const int t0 = bt0 - b * T_DIM;

  v8f acc[4] = {};

  for (int e0 = 0; e0 < E_DIM; e0 += 32) {
    // stage x: 128x32 f32 -> f16, 4 float4 per thread
#pragma unroll
    for (int c = 0; c < 4; ++c) {
      int lin = c * 1024 + tid * 4;
      int row = lin >> 5, col = lin & 31;
      float4 xv = *(const float4*)(x + (size_t)(bt0 + row) * E_DIM + e0 + col);
      v4h hv = {(_Float16)xv.x, (_Float16)xv.y, (_Float16)xv.z, (_Float16)xv.w};
      *(v4h*)&xs[row * 32 + col] = hv;
    }
    // stage W^T: wt[d][ec] = W[h][e0+ec][d]
    {
      int ec = tid >> 3;
      int d0 = (tid & 7) * 8;
      const float* wp = W + ((size_t)h * E_DIM + e0 + ec) * D_DIM + d0;
      float4 w0 = *(const float4*)wp;
      float4 w1 = *(const float4*)(wp + 4);
      wt[(d0 + 0) * 32 + ec] = (_Float16)w0.x;
      wt[(d0 + 1) * 32 + ec] = (_Float16)w0.y;
      wt[(d0 + 2) * 32 + ec] = (_Float16)w0.z;
      wt[(d0 + 3) * 32 + ec] = (_Float16)w0.w;
      wt[(d0 + 4) * 32 + ec] = (_Float16)w1.x;
      wt[(d0 + 5) * 32 + ec] = (_Float16)w1.y;
      wt[(d0 + 6) * 32 + ec] = (_Float16)w1.z;
      wt[(d0 + 7) * 32 + ec] = (_Float16)w1.w;
    }
    __syncthreads();
    const int wv = tid >> 5;
    v16h a = load_fragA(&xs[(wv * 16) * 32], 32);
#pragma unroll
    for (int nt = 0; nt < 4; ++nt)
      acc[nt] = wmma16(a, load_fragB(&wt[(nt * 16) * 32], 32), acc[nt]);
    __syncthreads();
  }

  const int lane = tid & 31, wv = tid >> 5;
  const int n = lane & 15, hi = lane >> 4;
#pragma unroll
  for (int nt = 0; nt < 4; ++nt) {
    int d = nt * 16 + n;
    float bb = bias[h * D_DIM + d];
#pragma unroll
    for (int j = 0; j < 8; ++j) {
      int t = t0 + wv * 16 + j + hi * 8;
      float val = acc[nt][j] + bb;
      if (!transpose_out)
        out[(((size_t)b * H_DIM + h) * T_DIM + t) * D_DIM + d] = (_Float16)val;
      else
        out[(((size_t)b * H_DIM + h) * D_DIM + d) * T_DIM + t] = (_Float16)val;
    }
  }
}

// ---------------------------------------------------------------------------
// Kernel 2: causal flash attention. Block = 128 query rows for one (b,h);
// 8 waves x 16-row strips. Online softmax, s-blocks of 32 keys.
// ---------------------------------------------------------------------------
__global__ __launch_bounds__(256) void flash_attn_kernel(
    const _Float16* __restrict__ q,   // [B,H,T,D]
    const _Float16* __restrict__ k,   // [B,H,T,D]
    const _Float16* __restrict__ vt,  // [B,H,D,T]
    _Float16* __restrict__ cat) {     // [B,T,H*D]
  __shared__ __align__(16) _Float16 kt[32 * 64];        // keys (row-major)
  __shared__ __align__(16) _Float16 vtile[64 * 32];     // V^T tile
  __shared__ __align__(16) _Float16 pbuf[8 * 16 * 32];  // per-wave P

  const int tid = threadIdx.x, lane = tid & 31, wv = tid >> 5;
  const int n = lane & 15, hi = lane >> 4;
  const int q0 = blockIdx.x * 128;
  const int bh = blockIdx.y;
  const int b = bh >> 4, h = bh & 15;
  const size_t base_kq = (size_t)bh * T_DIM * D_DIM;
  const size_t base_vt = (size_t)bh * D_DIM * T_DIM;
  const int strip = q0 + wv * 16;

  // Q fragments for this strip, kept in registers for the whole s-loop
  v16h qa0 = load_fragA(q + base_kq + (size_t)strip * D_DIM, D_DIM);
  v16h qa1 = load_fragA(q + base_kq + (size_t)strip * D_DIM + 32, D_DIM);

  v8f o0 = {}, o1 = {}, o2 = {}, o3 = {};
  float mrow[8], lrow[8];
#pragma unroll
  for (int j = 0; j < 8; ++j) { mrow[j] = -1e30f; lrow[j] = 0.0f; }
  const float sc = 0.125f * 1.44269504088896340736f;  // 1/sqrt(D) * log2(e)

  const int nsb = (q0 + 128) / 32;
  for (int sb = 0; sb < nsb; ++sb) {
    const int s0 = sb * 32;
    {  // cooperative staging
      int r = tid >> 3, c = (tid & 7) * 8;  // 32 rows x 64 d of K
      *(v8h*)&kt[r * 64 + c] =
          *(const v8h*)(k + base_kq + (size_t)(s0 + r) * D_DIM + c);
      int d = tid >> 2, c2 = (tid & 3) * 8;  // 64 rows x 32 s of V^T
      *(v8h*)&vtile[d * 32 + c2] =
          *(const v8h*)(vt + base_vt + (size_t)d * T_DIM + s0 + c2);
    }
    __syncthreads();

    if (s0 <= strip + 15) {  // block has at least one unmasked column
      v8f s[2];
#pragma unroll
      for (int nt = 0; nt < 2; ++nt) {
        v8f a = {};
        v16h b0 = load_fragB(&kt[(nt * 16) * 64], 64);
        v16h b1 = load_fragB(&kt[(nt * 16) * 64 + 32], 64);
        a = wmma16(qa0, b0, a);
        a = wmma16(qa1, b1, a);
        s[nt] = a;
      }
      float pr0[8], pr1[8];
#pragma unroll
      for (int j = 0; j < 8; ++j) {
        int t = strip + j + hi * 8;
        float sv0 = s[0][j] * sc, sv1 = s[1][j] * sc;
        if (s0 + n > t) sv0 = -1e30f;
        if (s0 + 16 + n > t) sv1 = -1e30f;
        float rmax = fmaxf(sv0, sv1);
#pragma unroll
        for (int msk = 8; msk >= 1; msk >>= 1)
          rmax = fmaxf(rmax, __shfl_xor(rmax, msk, 32));
        float mnew = fmaxf(mrow[j], rmax);
        float corr = exp2f(mrow[j] - mnew);
        float p0 = exp2f(sv0 - mnew), p1 = exp2f(sv1 - mnew);
        float rs = p0 + p1;
#pragma unroll
        for (int msk = 8; msk >= 1; msk >>= 1) rs += __shfl_xor(rs, msk, 32);
        lrow[j] = lrow[j] * corr + rs;
        mrow[j] = mnew;
        o0[j] *= corr; o1[j] *= corr; o2[j] *= corr; o3[j] *= corr;
        pr0[j] = p0; pr1[j] = p1;
      }
      // C-layout -> A-layout relayout of P through per-wave LDS
      _Float16* pb = &pbuf[wv * 16 * 32];
#pragma unroll
      for (int j = 0; j < 8; ++j) {
        pb[(j + hi * 8) * 32 + n] = (_Float16)pr0[j];
        pb[(j + hi * 8) * 32 + 16 + n] = (_Float16)pr1[j];
      }
      asm volatile("s_wait_dscnt 0" ::: "memory");  // LDS store->load hazard
      v16h pa = load_fragA(pb, 32);
      o0 = wmma16(pa, load_fragB(&vtile[0 * 16 * 32], 32), o0);
      o1 = wmma16(pa, load_fragB(&vtile[1 * 16 * 32], 32), o1);
      o2 = wmma16(pa, load_fragB(&vtile[2 * 16 * 32], 32), o2);
      o3 = wmma16(pa, load_fragB(&vtile[3 * 16 * 32], 32), o3);
    }
    __syncthreads();
  }

#pragma unroll
  for (int j = 0; j < 8; ++j) {
    float inv = 1.0f / lrow[j];
    int t = strip + j + hi * 8;
    _Float16* dst = cat + ((size_t)b * T_DIM + t) * E_DIM + h * D_DIM;
    dst[0 * 16 + n] = (_Float16)(o0[j] * inv);
    dst[1 * 16 + n] = (_Float16)(o1[j] * inv);
    dst[2 * 16 + n] = (_Float16)(o2[j] * inv);
    dst[3 * 16 + n] = (_Float16)(o3[j] * inv);
  }
}

// ---------------------------------------------------------------------------
// Kernel 3: out = cat(f16) @ Wp + bp, fp32 output. Tile 128x64 per block.
// ---------------------------------------------------------------------------
__global__ __launch_bounds__(256) void out_proj_kernel(
    const _Float16* __restrict__ cat, const float* __restrict__ Wp,
    const float* __restrict__ bp, float* __restrict__ out) {
  __shared__ __align__(16) _Float16 xs[128 * 32];
  __shared__ __align__(16) _Float16 wt[64 * 32];  // Wp^T tile: wt[nn][e]

  const int tid = threadIdx.x;
  const int r0 = blockIdx.x * 128;  // flattened b*T+t rows
  const int n0 = blockIdx.y * 64;   // output columns

  v8f acc[4] = {};
  for (int e0 = 0; e0 < E_DIM; e0 += 32) {
#pragma unroll
    for (int c = 0; c < 2; ++c) {  // stage cat tile, 2 x b128 per thread
      int lin = c * 2048 + tid * 8;
      int row = lin >> 5, col = lin & 31;
      *(v8h*)&xs[row * 32 + col] =
          *(const v8h*)(cat + (size_t)(r0 + row) * E_DIM + e0 + col);
    }
    {  // stage Wp^T: wt[nn][ec] = Wp[e0+ec][n0+nn]
      int ec = tid >> 3;
      int nb = (tid & 7) * 8;
      const float* wp = Wp + (size_t)(e0 + ec) * E_DIM + n0 + nb;
      float4 w0 = *(const float4*)wp;
      float4 w1 = *(const float4*)(wp + 4);
      wt[(nb + 0) * 32 + ec] = (_Float16)w0.x;
      wt[(nb + 1) * 32 + ec] = (_Float16)w0.y;
      wt[(nb + 2) * 32 + ec] = (_Float16)w0.z;
      wt[(nb + 3) * 32 + ec] = (_Float16)w0.w;
      wt[(nb + 4) * 32 + ec] = (_Float16)w1.x;
      wt[(nb + 5) * 32 + ec] = (_Float16)w1.y;
      wt[(nb + 6) * 32 + ec] = (_Float16)w1.z;
      wt[(nb + 7) * 32 + ec] = (_Float16)w1.w;
    }
    __syncthreads();
    const int wv = tid >> 5;
    v16h a = load_fragA(&xs[(wv * 16) * 32], 32);
#pragma unroll
    for (int nt = 0; nt < 4; ++nt)
      acc[nt] = wmma16(a, load_fragB(&wt[(nt * 16) * 32], 32), acc[nt]);
    __syncthreads();
  }

  const int lane = tid & 31, wv = tid >> 5;
  const int n = lane & 15, hi = lane >> 4;
#pragma unroll
  for (int nt = 0; nt < 4; ++nt) {
    float bb = bp[n0 + nt * 16 + n];
#pragma unroll
    for (int j = 0; j < 8; ++j) {
      int row = r0 + wv * 16 + j + hi * 8;
      out[(size_t)row * E_DIM + n0 + nt * 16 + n] = acc[nt][j] + bb;
    }
  }
}

extern "C" void kernel_launch(void* const* d_in, const int* in_sizes, int n_in,
                              void* d_out, int out_size, void* d_ws,
                              size_t ws_size, hipStream_t stream) {
  (void)in_sizes; (void)n_in; (void)out_size; (void)ws_size;
  const float* x  = (const float*)d_in[0];
  const float* Wq = (const float*)d_in[1];
  const float* bq = (const float*)d_in[2];
  const float* Wk = (const float*)d_in[3];
  const float* bk = (const float*)d_in[4];
  const float* Wv = (const float*)d_in[5];
  const float* bv = (const float*)d_in[6];
  const float* Wp = (const float*)d_in[7];
  const float* bp = (const float*)d_in[8];
  float* out = (float*)d_out;

  const size_t nQ = (size_t)B_DIM * H_DIM * T_DIM * D_DIM;  // 8M elems
  _Float16* qh  = (_Float16*)d_ws;
  _Float16* kh  = qh + nQ;
  _Float16* vth = kh + nQ;
  _Float16* cth = vth + nQ;  // [B,T,E] f16

  dim3 blk(256);
  dim3 g1(B_DIM * T_DIM / 128, H_DIM);
  qkv_proj_kernel<<<g1, blk, 0, stream>>>(x, Wq, bq, qh, 0);
  qkv_proj_kernel<<<g1, blk, 0, stream>>>(x, Wk, bk, kh, 0);
  qkv_proj_kernel<<<g1, blk, 0, stream>>>(x, Wv, bv, vth, 1);

  dim3 g2(T_DIM / 128, B_DIM * H_DIM);
  flash_attn_kernel<<<g2, blk, 0, stream>>>(qh, kh, vth, cth);

  dim3 g3(B_DIM * T_DIM / 128, E_DIM / 64);
  out_proj_kernel<<<g3, blk, 0, stream>>>(cth, Wp, bp, out);
}
